// MultiheadAttention_74028056314062
// MI455X (gfx1250) — matrix operand order repaired and checked by reading, not compiled, not run on hardware
//
#include <hip/hip_runtime.h>

// ---------------- problem constants ----------------
#define DK   1024
#define SEQ  2048
#define NB   2
#define NH   16
#define HD   64
#define MTOT (NB * SEQ)   // 4096 tokens

// ---------------- CDNA5 WMMA types ----------------
typedef __bf16 v16bf __attribute__((ext_vector_type(16)));
typedef float  v8f   __attribute__((ext_vector_type(8)));
typedef unsigned short ushort8 __attribute__((ext_vector_type(8)));
typedef unsigned int   v4u __attribute__((ext_vector_type(4)));
typedef unsigned int   v8u __attribute__((ext_vector_type(8)));
typedef unsigned short u16;

union Frag { v16bf v; ushort8 u[2]; };

// float -> bf16 bits, round-to-nearest-even
__device__ __forceinline__ u16 f2bf(float f) {
  unsigned u = __builtin_bit_cast(unsigned, f);
  return (u16)((u + 0x7FFFu + ((u >> 16) & 1u)) >> 16);
}

__device__ __forceinline__ v8f wmma_bf16(v16bf a, v16bf b, v8f c) {
  // D = A(16x32 bf16) * B(32x16 bf16) + C(16x16 f32)
  return __builtin_amdgcn_wmma_f32_16x16x32_bf16(
      false, a, false, b, (short)0, c, false, false);
}

// A-fragment 16x32: rows in lanes 0-15; half h: linear K = 16h + e
__device__ __forceinline__ v16bf load_frag_a(const u16* src, int row0, int k0,
                                             int ld, int lane) {
  const int m = lane & 15, h = lane >> 4;
  const u16* p = src + (size_t)(row0 + m) * ld + k0;
  Frag f;
  f.u[0] = *(const ushort8*)(p + 16 * h);
  f.u[1] = *(const ushort8*)(p + 16 * h + 8);
  return f.v;
}

// B-fragment 32x16: columns in lanes (n = lane&15), linear K = 16h + e.
// B[k][n] = src[(col0+n)*ld + k0 + k]
__device__ __forceinline__ v16bf load_frag_b(const u16* src, int col0, int k0,
                                             int ld, int lane) {
  const int n = lane & 15, h = lane >> 4;
  const u16* p = src + (size_t)(col0 + n) * ld + k0 + 16 * h;
  Frag f;
  f.u[0] = *(const ushort8*)p;
  f.u[1] = *(const ushort8*)(p + 8);
  return f.v;
}

// half-group (16 lane) row reductions for the WMMA C/D layout
__device__ __forceinline__ float row_max16(float x) {
  x = fmaxf(x, __shfl_xor(x, 1, 32));
  x = fmaxf(x, __shfl_xor(x, 2, 32));
  x = fmaxf(x, __shfl_xor(x, 4, 32));
  x = fmaxf(x, __shfl_xor(x, 8, 32));
  return x;
}
__device__ __forceinline__ float row_sum16(float x) {
  x += __shfl_xor(x, 1, 32);
  x += __shfl_xor(x, 2, 32);
  x += __shfl_xor(x, 4, 32);
  x += __shfl_xor(x, 8, 32);
  return x;
}

// ---------------- Tensor Data Mover: 2-D tile global -> LDS ----------------
// D# packed per CDNA5 ISA ch.8 (group0 + group1; VADDR2/3 = NULL => 2-D).
// tile_d0 = elements along contiguous dim, tile_d1 = rows, stride0 = row stride
// (elements). tensor dims == tile dims (tile fully in-bounds, no OOB clamping).
__device__ __forceinline__ void tdm_load_2d(unsigned lds_off, const u16* gptr,
                                            unsigned tile_d0, unsigned tile_d1,
                                            unsigned stride0) {
  unsigned long long ga = (unsigned long long)(size_t)gptr;
  v4u g0;
  g0.x = 1u;                                        // count=1, user descriptor
  g0.y = lds_off;                                   // lds_addr (bytes)
  g0.z = (unsigned)ga;                              // global_addr[31:0]
  g0.w = (unsigned)((ga >> 32) & 0x01ffffffu)       // global_addr[56:32]
         | (2u << 30);                              // type = 2 ("image")
  v8u g1;
  g1.s0 = (1u << 16);                               // wg_mask=0, data_size=2B
  g1.s1 = (tile_d0 & 0xffffu) << 16;                // tensor_dim0[15:0]
  g1.s2 = (tile_d0 >> 16) | ((tile_d1 & 0xffffu) << 16);   // td0 hi | td1 lo
  g1.s3 = (tile_d1 >> 16) | (tile_d0 << 16);        // td1 hi | tile_dim0
  g1.s4 = tile_d1 & 0xffffu;                        // tile_dim1 (tile_dim2=0)
  g1.s5 = stride0;                                  // tensor_dim0_stride[31:0]
  g1.s6 = 0;                                        // stride0 hi | stride1 lo
  g1.s7 = 0;
  asm volatile("tensor_load_to_lds %0, %1" :: "s"(g0), "s"(g1) : "memory");
}

__device__ __forceinline__ unsigned lds_off_u32(const void* p) {
  return (unsigned)(size_t)p;   // low 32 bits of generic addr = LDS byte offset
}

// ---------------- TDM double-buffered GEMM mainloop ----------------
// Block tile 64(M) x 128(N), 8 waves as 2x4 of 32x32 wave tiles, K-step 32.
// acc[0]=MN(0,0) acc[1]=(0,1) acc[2]=(1,0) acc[3]=(1,1)
__device__ __forceinline__ void gemm_tile_tdm(const u16* Abase, const u16* Wbase,
                                              int blk_m, int blk_n, int wave,
                                              int lane, u16* sA, u16* sB,
                                              v8f* acc) {
  const int lm  = (wave >> 2) * 32;   // wave row offset in block tile
  const int lnn = (wave & 3) * 32;    // wave col offset in block tile
  const unsigned offA = lds_off_u32(sA);
  const unsigned offB = lds_off_u32(sB);
  const u16* Ab = Abase + (size_t)blk_m * DK;
  const u16* Bb = Wbase + (size_t)blk_n * DK;

  if (wave == 0) {                    // prologue: fill buffer 0
    tdm_load_2d(offA, Ab, 32, 64, DK);
    tdm_load_2d(offB, Bb, 32, 128, DK);
  }
  int cur = 0;
  for (int it = 0; it < DK / 32; ++it) {
    __builtin_amdgcn_s_wait_tensorcnt(0);   // issuer waits; NOP for others
    __syncthreads();                        // buf[cur] ready; buf[cur^1] free
    if (wave == 0 && it + 1 < DK / 32) {    // prefetch next K-step via TDM
      int k0n = (it + 1) * 32;
      int nb = cur ^ 1;
      tdm_load_2d(offA + nb * (64 * 32 * 2),  Ab + k0n, 32, 64,  DK);
      tdm_load_2d(offB + nb * (128 * 32 * 2), Bb + k0n, 32, 128, DK);
    }
    const u16* As = sA + cur * (64 * 32);
    const u16* Bs = sB + cur * (128 * 32);
    v16bf fa0 = load_frag_a(As, lm, 0, 32, lane);
    v16bf fa1 = load_frag_a(As, lm + 16, 0, 32, lane);
    v16bf fb0 = load_frag_b(Bs, lnn, 0, 32, lane);
    v16bf fb1 = load_frag_b(Bs, lnn + 16, 0, 32, lane);
    acc[0] = wmma_bf16(fa0, fb0, acc[0]);
    acc[1] = wmma_bf16(fa0, fb1, acc[1]);
    acc[2] = wmma_bf16(fa1, fb0, acc[2]);
    acc[3] = wmma_bf16(fa1, fb1, acc[3]);
    cur ^= 1;
  }
}

// ---------------- kernel 0: f32 -> bf16 conversion ----------------
__global__ __launch_bounds__(256) void cvt_kernel(const float* __restrict__ in,
                                                  u16* __restrict__ out, int n) {
  int i = (blockIdx.x * 256 + threadIdx.x) * 4;
  if (i + 3 < n) {
    float4 f = *(const float4*)(in + i);
    out[i + 0] = f2bf(f.x);
    out[i + 1] = f2bf(f.y);
    out[i + 2] = f2bf(f.z);
    out[i + 3] = f2bf(f.w);
  }
}

// ---------------- kernel 1: fused QKV projection (grid.z = 0/1/2) ------------
__global__ __launch_bounds__(256) void qkv_proj_kernel(
    const u16* __restrict__ xb,
    const u16* __restrict__ wqb, const u16* __restrict__ wkb,
    const u16* __restrict__ wvb,
    const float* __restrict__ bq, const float* __restrict__ bk,
    const float* __restrict__ bv,
    u16* __restrict__ Qb, u16* __restrict__ Kb, u16* __restrict__ Vtb) {
  __shared__ __align__(16) u16 sA[2 * 64 * 32];    // 8 KiB
  __shared__ __align__(16) u16 sB[2 * 128 * 32];   // 16 KiB
  const int lane = threadIdx.x & 31;
  const int wave = threadIdx.x >> 5;
  const int z = blockIdx.z;
  const int blk_m = blockIdx.x * 64;
  const int blk_n = blockIdx.y * 128;
  const u16* W = (z == 0) ? wqb : (z == 1) ? wkb : wvb;
  const float* bias = (z == 0) ? bq : (z == 1) ? bk : bv;

  v8f acc[4] = {{}, {}, {}, {}};
  gemm_tile_tdm(xb, W, blk_m, blk_n, wave, lane, sA, sB, acc);

  const int h = lane >> 4, ln = lane & 15;
  const int m0 = blk_m + (wave >> 2) * 32;
  const int n0 = blk_n + (wave & 3) * 32;
#pragma unroll
  for (int i = 0; i < 2; ++i) {
#pragma unroll
    for (int v = 0; v < 8; ++v) {
      int gm = m0 + i * 16 + 8 * h + v;
      int bidx = gm >> 11;            // / SEQ
      int s = gm & (SEQ - 1);
#pragma unroll
      for (int j = 0; j < 2; ++j) {
        int gn = n0 + j * 16 + ln;
        float val = acc[i * 2 + j][v] + bias[gn];
        int hh = gn >> 6, dd = gn & 63;
        if (z == 0) {                 // Q, pre-scaled by 1/sqrt(64)
          Qb[((size_t)(bidx * NH + hh) * SEQ + s) * HD + dd] = f2bf(val * 0.125f);
        } else if (z == 1) {          // K
          Kb[((size_t)(bidx * NH + hh) * SEQ + s) * HD + dd] = f2bf(val);
        } else {                      // V transposed (d-major) for P*V B-frags
          Vtb[((size_t)(bidx * NH + hh) * HD + dd) * SEQ + s] = f2bf(val);
        }
      }
    }
  }
}

// ---------------- kernel 2: causal flash attention ----------------
// One wave: 16 queries x full head (hd=64); 8 waves / block = 128 queries.
__global__ __launch_bounds__(256) void flash_attn_kernel(
    const u16* __restrict__ Qb, const u16* __restrict__ Kb,
    const u16* __restrict__ Vtb, u16* __restrict__ Ctxb) {
  __shared__ __align__(16) u16 plds[8 * 16 * 32];   // per-wave P staging (8KB)
  const int lane = threadIdx.x & 31;
  const int wave = threadIdx.x >> 5;
  const int bh = blockIdx.y;                        // b*NH + h
  const int q0 = blockIdx.x * 128 + wave * 16;
  const u16* Qh = Qb + (size_t)bh * SEQ * HD;
  const u16* Kh = Kb + (size_t)bh * SEQ * HD;
  const u16* Vh = Vtb + (size_t)bh * HD * SEQ;
  const int h = lane >> 4, ln = lane & 15;

  v16bf qa0 = load_frag_a(Qh, q0, 0, HD, lane);     // Q pre-scaled by 1/sqrt(hd)
  v16bf qa1 = load_frag_a(Qh, q0, 32, HD, lane);

  v8f o0 = {}, o1 = {}, o2 = {}, o3 = {};           // 16x64 f32 context accum
  float mstate[8], lstate[8];
#pragma unroll
  for (int v = 0; v < 8; ++v) { mstate[v] = -1e30f; lstate[v] = 0.f; }

  u16* pl = &plds[wave * 512];
  const int kend = q0 + 16;                         // causal

  for (int k0 = 0; k0 < kend; k0 += 32) {
    if (k0 + 32 < kend) {                           // global_prefetch next K tile
      __builtin_prefetch(Kh + (size_t)(k0 + 32 + ln) * HD, 0, 1);
      __builtin_prefetch(Kh + (size_t)(k0 + 48 + ln) * HD, 0, 1);
    }
    v8f s0 = {}, s1 = {};
    s0 = wmma_bf16(qa0, load_frag_b(Kh, k0, 0, HD, lane), s0);
    s0 = wmma_bf16(qa1, load_frag_b(Kh, k0, 32, HD, lane), s0);
    s1 = wmma_bf16(qa0, load_frag_b(Kh, k0 + 16, 0, HD, lane), s1);
    s1 = wmma_bf16(qa1, load_frag_b(Kh, k0 + 16, 32, HD, lane), s1);

#pragma unroll
    for (int v = 0; v < 8; ++v) {                   // causal mask
      int qi = q0 + 8 * h + v;
      if (k0 + ln > qi)      s0[v] = -3.0e38f;
      if (k0 + 16 + ln > qi) s1[v] = -3.0e38f;
    }

    float alpha[8];
#pragma unroll
    for (int v = 0; v < 8; ++v) {                   // online softmax
      float r = row_max16(fmaxf(s0[v], s1[v]));
      float mn = fmaxf(mstate[v], r);
      alpha[v] = __expf(mstate[v] - mn);
      mstate[v] = mn;
      float p0 = __expf(s0[v] - mn);
      float p1 = __expf(s1[v] - mn);
      s0[v] = p0; s1[v] = p1;
      lstate[v] = lstate[v] * alpha[v] + row_sum16(p0 + p1);
    }
#pragma unroll
    for (int v = 0; v < 8; ++v) {
      o0[v] *= alpha[v]; o1[v] *= alpha[v];
      o2[v] *= alpha[v]; o3[v] *= alpha[v];
    }

    // P: C-layout -> LDS (bf16) -> A-fragment layout (wave-private, in-order)
#pragma unroll
    for (int v = 0; v < 8; ++v) {
      int r = 8 * h + v;
      pl[r * 32 + ln]      = f2bf(s0[v]);
      pl[r * 32 + 16 + ln] = f2bf(s1[v]);
    }
    v16bf pa = load_frag_a(pl, 0, 0, 32, lane);

    o0 = wmma_bf16(pa, load_frag_b(Vh, 0,  k0, SEQ, lane), o0);
    o1 = wmma_bf16(pa, load_frag_b(Vh, 16, k0, SEQ, lane), o1);
    o2 = wmma_bf16(pa, load_frag_b(Vh, 32, k0, SEQ, lane), o2);
    o3 = wmma_bf16(pa, load_frag_b(Vh, 48, k0, SEQ, lane), o3);
  }

  const int b = bh >> 4, hh = bh & 15;
#pragma unroll
  for (int v = 0; v < 8; ++v) {
    int r = q0 + 8 * h + v;
    float inv = 1.0f / lstate[v];
    size_t base = ((size_t)(b * SEQ + r)) * DK + hh * HD + ln;
    Ctxb[base]      = f2bf(o0[v] * inv);
    Ctxb[base + 16] = f2bf(o1[v] * inv);
    Ctxb[base + 32] = f2bf(o2[v] * inv);
    Ctxb[base + 48] = f2bf(o3[v] * inv);
  }
}

// ---------------- kernel 3: output projection (f32 out) ----------------
__global__ __launch_bounds__(256) void out_proj_kernel(
    const u16* __restrict__ ctxb, const u16* __restrict__ wob,
    const float* __restrict__ bo, float* __restrict__ out) {
  __shared__ __align__(16) u16 sA[2 * 64 * 32];
  __shared__ __align__(16) u16 sB[2 * 128 * 32];
  const int lane = threadIdx.x & 31;
  const int wave = threadIdx.x >> 5;
  const int blk_m = blockIdx.x * 64;
  const int blk_n = blockIdx.y * 128;

  v8f acc[4] = {{}, {}, {}, {}};
  gemm_tile_tdm(ctxb, wob, blk_m, blk_n, wave, lane, sA, sB, acc);

  const int h = lane >> 4, ln = lane & 15;
  const int m0 = blk_m + (wave >> 2) * 32;
  const int n0 = blk_n + (wave & 3) * 32;
#pragma unroll
  for (int i = 0; i < 2; ++i) {
#pragma unroll
    for (int v = 0; v < 8; ++v) {
      int gm = m0 + i * 16 + 8 * h + v;
      int gn = n0 + ln;
      out[(size_t)gm * DK + gn]      = acc[i * 2 + 0][v] + bo[gn];
      out[(size_t)gm * DK + gn + 16] = acc[i * 2 + 1][v] + bo[gn + 16];
    }
  }
}

// ---------------- launcher ----------------
extern "C" void kernel_launch(void* const* d_in, const int* in_sizes, int n_in,
                              void* d_out, int out_size, void* d_ws, size_t ws_size,
                              hipStream_t stream) {
  (void)in_sizes; (void)n_in; (void)out_size; (void)ws_size;
  const float* x  = (const float*)d_in[0];
  const float* Wq = (const float*)d_in[1];
  const float* bq = (const float*)d_in[2];
  const float* Wk = (const float*)d_in[3];
  const float* bk = (const float*)d_in[4];
  const float* Wv = (const float*)d_in[5];
  const float* bv = (const float*)d_in[6];
  const float* Wo = (const float*)d_in[7];
  const float* bo = (const float*)d_in[8];
  float* out = (float*)d_out;

  // workspace layout; total 48 MiB (fully L2-resident on MI455X)
  char* ws = (char*)d_ws;
  u16* xb   = (u16*)(ws + 0);
  u16* wqb  = (u16*)(ws + (8u  << 20));
  u16* wkb  = (u16*)(ws + (10u << 20));
  u16* wvb  = (u16*)(ws + (12u << 20));
  u16* wob  = (u16*)(ws + (14u << 20));
  u16* Qb   = (u16*)(ws + (16u << 20)); // (B,H,S,hd)
  u16* Kb   = (u16*)(ws + (24u << 20)); // (B,H,S,hd)
  u16* Vtb  = (u16*)(ws + (32u << 20)); // (B,H,hd,S)
  u16* Ctxb = (u16*)(ws + (40u << 20)); // (B,S,D)

  const int nx = MTOT * DK;   // 4,194,304
  const int nw = DK * DK;     // 1,048,576
  cvt_kernel<<<nx / 1024, 256, 0, stream>>>(x,  xb,  nx);
  cvt_kernel<<<nw / 1024, 256, 0, stream>>>(Wq, wqb, nw);
  cvt_kernel<<<nw / 1024, 256, 0, stream>>>(Wk, wkb, nw);
  cvt_kernel<<<nw / 1024, 256, 0, stream>>>(Wv, wvb, nw);
  cvt_kernel<<<nw / 1024, 256, 0, stream>>>(Wo, wob, nw);

  qkv_proj_kernel<<<dim3(MTOT / 64, DK / 128, 3), 256, 0, stream>>>(
      xb, wqb, wkb, wvb, bq, bk, bv, Qb, Kb, Vtb);

  flash_attn_kernel<<<dim3(SEQ / 128, NB * NH), 256, 0, stream>>>(Qb, Kb, Vtb, Ctxb);

  out_proj_kernel<<<dim3(MTOT / 64, DK / 128), 256, 0, stream>>>(Ctxb, wob, bo, out);
}
